// MultiHeadAttention_82806969467488
// MI455X (gfx1250) — matrix operand order repaired and checked by reading, not compiled, not run on hardware
//
#include <hip/hip_runtime.h>

#define B_ 2
#define S_ 4096
#define D_ 512
#define H_ 8
#define DH 64

typedef __attribute__((ext_vector_type(16))) _Float16 v16h;
typedef __attribute__((ext_vector_type(8)))  _Float16 v8h;
typedef __attribute__((ext_vector_type(8)))  float    v8f;
typedef __attribute__((ext_vector_type(4)))  float    v4f;
typedef __attribute__((ext_vector_type(4)))  int      v4i;

// ---- gfx1250 async global->LDS path (guarded; safe fallback otherwise) ----
#if defined(__has_builtin)
#if __has_builtin(__builtin_amdgcn_global_load_async_to_lds_b128)
#define USE_ASYNC_LDS 1
#endif
#endif
#ifndef USE_ASYNC_LDS
#define USE_ASYNC_LDS 0
#endif

typedef __attribute__((address_space(1))) v4i AS1v4i;  // global int4
typedef __attribute__((address_space(3))) v4i AS3v4i;  // LDS int4

#if USE_ASYNC_LDS
static __device__ __forceinline__ void async_cp16(const _Float16* g,
                                                  const _Float16* l) {
  // 16 bytes global -> LDS, tracked by ASYNCcnt. Generic->AS casts via
  // integers: AS1 ptr == 64-bit VA; AS3 ptr == low 32 bits of flat LDS addr.
  __builtin_amdgcn_global_load_async_to_lds_b128(
      (AS1v4i*)(size_t)(const void*)g,
      (AS3v4i*)(unsigned)(size_t)(const void*)l, 0, 0);
}
static __device__ __forceinline__ void wait_async0() {
#if __has_builtin(__builtin_amdgcn_s_wait_asynccnt)
  __builtin_amdgcn_s_wait_asynccnt(0);
#else
  __asm volatile("s_wait_asynccnt 0x0" ::: "memory");
#endif
}
#endif

static __device__ __forceinline__ v16h frag_cat(v8h lo, v8h hi) {
  v16h r;
#pragma unroll
  for (int i = 0; i < 8; ++i) { r[i] = lo[i]; r[i + 8] = hi[i]; }
  return r;
}
// A-matrix f16 16x32 fragment: per-lane two contiguous 8-half runs at p, p+16
static __device__ __forceinline__ v16h load_afrag(const _Float16* p) {
  return frag_cat(*(const v8h*)p, *(const v8h*)(p + 16));
}
// B-matrix f16 32x16 fragment: per-lane two contiguous 8-half runs at p, p+16
static __device__ __forceinline__ v16h load_bfrag(const _Float16* p) {
  return frag_cat(*(const v8h*)p, *(const v8h*)(p + 16));
}
// A fragment sourced from f32 memory, converted to f16
static __device__ __forceinline__ v16h load_afrag_f32(const float* p) {
  v4f a0 = *(const v4f*)p,        a1 = *(const v4f*)(p + 4);
  v4f a2 = *(const v4f*)(p + 16), a3 = *(const v4f*)(p + 20);
  v16h r;
#pragma unroll
  for (int i = 0; i < 4; ++i) {
    r[i]      = (_Float16)a0[i];
    r[i + 4]  = (_Float16)a1[i];
    r[i + 8]  = (_Float16)a2[i];
    r[i + 12] = (_Float16)a3[i];
  }
  return r;
}
static __device__ __forceinline__ v8f wmma16(v16h a, v16h b, v8f c) {
  return __builtin_amdgcn_wmma_f32_16x16x32_f16(false, a, false, b,
                                                (short)0, c, false, false);
}

// ---------------------------------------------------------------------------
// Kernel 1: convert + transpose four DxD weights to f16 (WT[n*D+k] = W[k*D+n])
// ---------------------------------------------------------------------------
__global__ __launch_bounds__(256) void cvtw_kernel(
    const float* __restrict__ Wq, const float* __restrict__ Wk,
    const float* __restrict__ Wv, const float* __restrict__ Wo,
    _Float16* __restrict__ WT) {
  int i = blockIdx.x * 256 + threadIdx.x;  // 0 .. 4*512*512-1
  int w = i >> 18;
  int rem = i & 262143;
  int n = rem >> 9, k = rem & 511;
  const float* W = (w == 0) ? Wq : (w == 1) ? Wk : (w == 2) ? Wv : Wo;
  WT[(size_t)w * 262144 + n * 512 + k] = (_Float16)W[k * 512 + n];
}

// ---------------------------------------------------------------------------
// Kernel 2: Q/K/V projections. Per wave: 32x64 output tile (2 A-frags shared
// against 4 B-frags -> 8 WMMA per k-step). Q pre-scaled by 1/sqrt(dh)=0.125.
// V stored transposed [B,H,dh,S] so attention PV B-fragments are contiguous.
// ---------------------------------------------------------------------------
__global__ __launch_bounds__(256) void qkv_proj_kernel(
    const float* __restrict__ Xq, const float* __restrict__ Xk,
    const float* __restrict__ Xv, const _Float16* __restrict__ WT,
    const float* __restrict__ bq, const float* __restrict__ bk,
    const float* __restrict__ bv,
    _Float16* __restrict__ Qh, _Float16* __restrict__ Kh,
    _Float16* __restrict__ Vt) {
  int wave = blockIdx.x * 8 + (threadIdx.x >> 5);
  int lane = threadIdx.x & 31;
  int which = wave >> 11;                // /2048: 0=Q, 1=K, 2=V
  int t = wave & 2047;
  int m0 = (t >> 3) << 5;                // 32-row tile base
  int n0 = (t & 7) << 6;                 // 64-wide col group base
  const float* X = (which == 0) ? Xq : (which == 1) ? Xk : Xv;
  const _Float16* W = WT + (size_t)which * 262144;
  const float* bias = (which == 0) ? bq : (which == 1) ? bk : bv;
  int mlo = lane & 15, khalf = lane >> 4;

  v8f acc[2][4];
#pragma unroll
  for (int mi = 0; mi < 2; ++mi)
#pragma unroll
    for (int tt = 0; tt < 4; ++tt)
#pragma unroll
      for (int r = 0; r < 8; ++r) acc[mi][tt][r] = 0.f;

  const float* arow0 = X + (size_t)(m0 + mlo) * D_;
  const float* arow1 = arow0 + (size_t)16 * D_;
  for (int k0 = 0; k0 < D_; k0 += 32) {
    v16h a0 = load_afrag_f32(arow0 + k0 + khalf * 8);
    v16h a1 = load_afrag_f32(arow1 + k0 + khalf * 8);
#pragma unroll
    for (int tt = 0; tt < 4; ++tt) {
      v16h bfr = load_bfrag(W + (size_t)(n0 + tt * 16 + mlo) * D_ + k0 +
                            khalf * 8);
      acc[0][tt] = wmma16(a0, bfr, acc[0][tt]);
      acc[1][tt] = wmma16(a1, bfr, acc[1][tt]);
    }
  }

#pragma unroll
  for (int tt = 0; tt < 4; ++tt) {
    int col = n0 + tt * 16 + mlo;
    float bcol = bias[col];
    int h = col >> 6, d = col & 63;
#pragma unroll
    for (int mi = 0; mi < 2; ++mi)
#pragma unroll
      for (int r = 0; r < 8; ++r) {
        int row = m0 + mi * 16 + r + (khalf << 3);
        int b = row >> 12, s = row & 4095;
        float v = acc[mi][tt][r] + bcol;
        if (which == 0)
          Qh[((size_t)(b * H_ + h) * S_ + s) * DH + d] =
              (_Float16)(v * 0.125f);
        else if (which == 1)
          Kh[((size_t)(b * H_ + h) * S_ + s) * DH + d] = (_Float16)v;
        else
          Vt[((size_t)(b * H_ + h) * DH + d) * S_ + s] = (_Float16)v;
      }
  }
}

// ---------------------------------------------------------------------------
// Kernel 3: flash attention. Block = 8 waves sharing one (b,h); 128 queries
// per block. K/V key-blocks of 64 staged cooperatively into LDS (async,
// double-buffered when available), then consumed 8x by the waves' WMMAs.
// Online softmax; P transposed C->A layout via per-wave LDS tile.
// ---------------------------------------------------------------------------
__global__ __launch_bounds__(256) void attn_kernel(
    const _Float16* __restrict__ Qh, const _Float16* __restrict__ Kh,
    const _Float16* __restrict__ Vt, const int* __restrict__ mask,
    _Float16* __restrict__ Ctx) {
#if USE_ASYNC_LDS
  __shared__ _Float16 Ksm[2 * 64 * 64];   // double-buffered 8KB tiles
  __shared__ _Float16 Vsm[2 * 64 * 64];
#else
  __shared__ _Float16 Ksm[64 * 64];
  __shared__ _Float16 Vsm[64 * 64];
#endif
  __shared__ _Float16 Psm[8 * 16 * 64];   // per-wave P tiles

  int wv = threadIdx.x >> 5, lane = threadIdx.x & 31;
  int tid = threadIdx.x;
  int bh = blockIdx.x >> 5;               // (b*H + h) in 0..15
  int q0 = (blockIdx.x & 31) * 128 + wv * 16;
  int b = bh >> 3, h = bh & 7;
  const _Float16* Qb = Qh + (size_t)bh * S_ * DH;
  const _Float16* Kb = Kh + (size_t)bh * S_ * DH;
  const _Float16* Vb = Vt + (size_t)bh * DH * S_;
  const int* mb = mask + b * S_;
  int mlo = lane & 15, khalf = lane >> 4;
  _Float16* Pw = Psm + wv * 1024;

  // staging assignment: 64 rows x 4 segments of 16 halves (32B/thread/tile)
  int srow = tid >> 2, sseg = (tid & 3) * 16;

  // resident Q A-fragments (pre-scaled by 0.125 in projection)
  v16h aq[2];
#pragma unroll
  for (int c = 0; c < 2; ++c)
    aq[c] = load_afrag(Qb + (size_t)(q0 + mlo) * DH + c * 32 + khalf * 8);

  float mrow[8], lrow[8];
  v8f o[4];
#pragma unroll
  for (int r = 0; r < 8; ++r) { mrow[r] = -__builtin_inff(); lrow[r] = 0.f; }
#pragma unroll
  for (int dt = 0; dt < 4; ++dt)
#pragma unroll
    for (int r = 0; r < 8; ++r) o[dt][r] = 0.f;

#if USE_ASYNC_LDS
  // ---- stage block 0 into buffer 0 ----
  {
    const _Float16* gk = Kb + (size_t)srow * DH + sseg;
    const _Float16* gv = Vb + (size_t)srow * S_ + sseg;
    _Float16* lk = Ksm + srow * 64 + sseg;
    _Float16* lv = Vsm + srow * 64 + sseg;
    async_cp16(gk, lk);
    async_cp16(gk + 8, lk + 8);
    async_cp16(gv, lv);
    async_cp16(gv + 8, lv + 8);
  }
  wait_async0();
  __syncthreads();
#else
  // ---- preload block 0 into registers ----
  v8h kr0, kr1, vr0, vr1;
  {
    const _Float16* gk = Kb + (size_t)srow * DH + sseg;
    const _Float16* gv = Vb + (size_t)srow * S_ + sseg;
    kr0 = *(const v8h*)gk;  kr1 = *(const v8h*)(gk + 8);
    vr0 = *(const v8h*)gv;  vr1 = *(const v8h*)(gv + 8);
  }
#endif

  for (int kb = 0, it = 0; kb < S_; kb += 64, ++it) {
#if USE_ASYNC_LDS
    const _Float16* Kl = Ksm + (it & 1) * 4096;
    const _Float16* Vl = Vsm + (it & 1) * 4096;
    if (kb + 64 < S_) {  // stage next block into other buffer
      const _Float16* gk = Kb + (size_t)(kb + 64 + srow) * DH + sseg;
      const _Float16* gv = Vb + (size_t)srow * S_ + kb + 64 + sseg;
      _Float16* lk = Ksm + ((it + 1) & 1) * 4096 + srow * 64 + sseg;
      _Float16* lv = Vsm + ((it + 1) & 1) * 4096 + srow * 64 + sseg;
      async_cp16(gk, lk);
      async_cp16(gk + 8, lk + 8);
      async_cp16(gv, lv);
      async_cp16(gv + 8, lv + 8);
    }
#else
    const _Float16* Kl = Ksm;
    const _Float16* Vl = Vsm;
    __syncthreads();  // prior block's readers done
    {
      _Float16* lk = Ksm + srow * 64 + sseg;
      _Float16* lv = Vsm + srow * 64 + sseg;
      *(v8h*)lk = kr0;       *(v8h*)(lk + 8) = kr1;
      *(v8h*)lv = vr0;       *(v8h*)(lv + 8) = vr1;
    }
    __syncthreads();  // tiles staged
    if (kb + 64 < S_) {  // prefetch next block into registers
      const _Float16* gk = Kb + (size_t)(kb + 64 + srow) * DH + sseg;
      const _Float16* gv = Vb + (size_t)srow * S_ + kb + 64 + sseg;
      kr0 = *(const v8h*)gk;  kr1 = *(const v8h*)(gk + 8);
      vr0 = *(const v8h*)gv;  vr1 = *(const v8h*)(gv + 8);
    }
#endif

    // ---- scores from LDS K tile: 4 key tiles x (dh=64 -> 2 WMMA) ----
    v8f sc[4];
#pragma unroll
    for (int t = 0; t < 4; ++t) {
#pragma unroll
      for (int r = 0; r < 8; ++r) sc[t][r] = 0.f;
      const _Float16* kp = Kl + (t * 16 + mlo) * 64 + khalf * 8;
#pragma unroll
      for (int c = 0; c < 2; ++c)
        sc[t] = wmma16(aq[c], load_bfrag(kp + c * 32), sc[t]);
    }

    // ---- key mask (per column) ----
#pragma unroll
    for (int t = 0; t < 4; ++t) {
      int mv = mb[kb + t * 16 + mlo];
#pragma unroll
      for (int r = 0; r < 8; ++r) sc[t][r] = (mv != 0) ? sc[t][r] : -1e9f;
    }

    // ---- online softmax: row max across 16-lane halves ----
    float mnew[8], alpha[8];
#pragma unroll
    for (int r = 0; r < 8; ++r) {
      float tm = sc[0][r];
#pragma unroll
      for (int t = 1; t < 4; ++t) tm = fmaxf(tm, sc[t][r]);
#pragma unroll
      for (int off = 1; off < 16; off <<= 1)
        tm = fmaxf(tm, __shfl_xor(tm, off, 16));
      mnew[r] = fmaxf(mrow[r], tm);
      alpha[r] = __expf(mrow[r] - mnew[r]);
      mrow[r] = mnew[r];
      lrow[r] *= alpha[r];
    }
#pragma unroll
    for (int dt = 0; dt < 4; ++dt)
#pragma unroll
      for (int r = 0; r < 8; ++r) o[dt][r] *= alpha[r];

    // ---- P = exp(s - m), row sums ----
#pragma unroll
    for (int r = 0; r < 8; ++r) {
      float rs = 0.f;
#pragma unroll
      for (int t = 0; t < 4; ++t) {
        float p = __expf(sc[t][r] - mnew[r]);
        sc[t][r] = p;
        rs += p;
      }
#pragma unroll
      for (int off = 1; off < 16; off <<= 1) rs += __shfl_xor(rs, off, 16);
      lrow[r] += rs;
    }

    // ---- C-layout -> A-layout transpose via per-wave LDS tile ----
#pragma unroll
    for (int t = 0; t < 4; ++t)
#pragma unroll
      for (int r = 0; r < 8; ++r)
        Pw[(r + (khalf << 3)) * 64 + t * 16 + mlo] = (_Float16)sc[t][r];
    __asm volatile("s_wait_dscnt 0x0" ::: "memory");  // same-wave LDS RAW
    v16h ap[2];
#pragma unroll
    for (int c = 0; c < 2; ++c)
      ap[c] = load_afrag(Pw + mlo * 64 + c * 32 + khalf * 8);

    // ---- O += P @ V from LDS V tile ----
#pragma unroll
    for (int dt = 0; dt < 4; ++dt) {
      const _Float16* vp = Vl + (dt * 16 + mlo) * 64 + khalf * 8;
#pragma unroll
      for (int c = 0; c < 2; ++c)
        o[dt] = wmma16(ap[c], load_bfrag(vp + c * 32), o[dt]);
    }

#if USE_ASYNC_LDS
    if (kb + 64 < S_) {
      wait_async0();     // own async copies for next buffer complete
      __syncthreads();   // everyone's copies complete + readers done
    }
#endif
  }

  // ---- normalize and store context [B,S,D] f16 ----
#pragma unroll
  for (int r = 0; r < 8; ++r) lrow[r] = 1.f / lrow[r];
  size_t rowbase = (size_t)b * S_ + q0;
#pragma unroll
  for (int dt = 0; dt < 4; ++dt) {
    int col = h * DH + dt * 16 + mlo;
#pragma unroll
    for (int r = 0; r < 8; ++r) {
      int m = r + (khalf << 3);
      Ctx[(rowbase + m) * D_ + col] = (_Float16)(o[dt][r] * lrow[r]);
    }
  }
}

// ---------------------------------------------------------------------------
// Kernel 4: output projection  out = Ctx(f16) @ Wo + bo -> f32 (32x64 tiles)
// ---------------------------------------------------------------------------
__global__ __launch_bounds__(256) void out_proj_kernel(
    const _Float16* __restrict__ Ctx, const _Float16* __restrict__ WoT,
    const float* __restrict__ bo, float* __restrict__ out) {
  int wave = blockIdx.x * 8 + (threadIdx.x >> 5);
  int lane = threadIdx.x & 31;
  int m0 = (wave >> 3) << 5;
  int n0 = (wave & 7) << 6;
  int mlo = lane & 15, khalf = lane >> 4;

  v8f acc[2][4];
#pragma unroll
  for (int mi = 0; mi < 2; ++mi)
#pragma unroll
    for (int tt = 0; tt < 4; ++tt)
#pragma unroll
      for (int r = 0; r < 8; ++r) acc[mi][tt][r] = 0.f;

  const _Float16* arow0 = Ctx + (size_t)(m0 + mlo) * D_;
  const _Float16* arow1 = arow0 + (size_t)16 * D_;
  for (int k0 = 0; k0 < D_; k0 += 32) {
    v16h a0 = load_afrag(arow0 + k0 + khalf * 8);
    v16h a1 = load_afrag(arow1 + k0 + khalf * 8);
#pragma unroll
    for (int tt = 0; tt < 4; ++tt) {
      v16h bfr = load_bfrag(WoT + (size_t)(n0 + tt * 16 + mlo) * D_ + k0 +
                            khalf * 8);
      acc[0][tt] = wmma16(a0, bfr, acc[0][tt]);
      acc[1][tt] = wmma16(a1, bfr, acc[1][tt]);
    }
  }

#pragma unroll
  for (int tt = 0; tt < 4; ++tt) {
    int col = n0 + tt * 16 + mlo;
    float bcol = bo[col];
#pragma unroll
    for (int mi = 0; mi < 2; ++mi)
#pragma unroll
      for (int r = 0; r < 8; ++r) {
        int row = m0 + mi * 16 + r + (khalf << 3);
        out[(size_t)row * D_ + col] = acc[mi][tt][r] + bcol;
      }
  }
}

// ---------------------------------------------------------------------------
extern "C" void kernel_launch(void* const* d_in, const int* in_sizes, int n_in,
                              void* d_out, int out_size, void* d_ws,
                              size_t ws_size, hipStream_t stream) {
  const float* query  = (const float*)d_in[0];
  const float* key_in = (const float*)d_in[1];
  const float* value  = (const float*)d_in[2];
  const int*   mask   = (const int*)d_in[3];
  const float* Wq = (const float*)d_in[4];
  const float* bq = (const float*)d_in[5];
  const float* Wk = (const float*)d_in[6];
  const float* bk = (const float*)d_in[7];
  const float* Wv = (const float*)d_in[8];
  const float* bv = (const float*)d_in[9];
  const float* Wo = (const float*)d_in[10];
  const float* bo = (const float*)d_in[11];

  // workspace layout (f16): 2MB WT(x4) | 8MB Qh | 8MB Kh | 8MB Vt | 8MB Ctx
  char* ws = (char*)d_ws;
  _Float16* WT  = (_Float16*)(ws);
  _Float16* Qh  = (_Float16*)(ws + (size_t)(2  << 20));
  _Float16* Kh  = (_Float16*)(ws + (size_t)(10 << 20));
  _Float16* Vt  = (_Float16*)(ws + (size_t)(18 << 20));
  _Float16* Ctx = (_Float16*)(ws + (size_t)(26 << 20));

  cvtw_kernel<<<4096, 256, 0, stream>>>(Wq, Wk, Wv, Wo, WT);
  qkv_proj_kernel<<<768, 256, 0, stream>>>(query, key_in, value, WT,
                                           bq, bk, bv, Qh, Kh, Vt);
  attn_kernel<<<512, 256, 0, stream>>>(Qh, Kh, Vt, mask, Ctx);
  out_proj_kernel<<<256, 256, 0, stream>>>(Ctx, WT + (size_t)3 * 262144, bo,
                                           (float*)d_out);
}
